// Classifier_v2_backup_29703993819343
// MI455X (gfx1250) — compile-verified, matrix operand order checked by reading, not run on hardware
//
#include <hip/hip_runtime.h>

typedef __attribute__((ext_vector_type(16))) _Float16 v16h;
typedef __attribute__((ext_vector_type(8)))  _Float16 h8;
typedef __attribute__((ext_vector_type(8)))  float    v8f;

#define NN 100000
#define NE 3200000
#define NG 32
#define D  128
#define NC 10

// ---------------- utility ----------------
__global__ void zero_f32(float* __restrict__ p, size_t n) {
  size_t i = (size_t)blockIdx.x * blockDim.x + threadIdx.x;
  size_t stride = (size_t)gridDim.x * blockDim.x;
  for (; i < n; i += stride) p[i] = 0.0f;
}

// degrees: odeg[src[e]] += 1, ideg[dst[e]] += 1
__global__ void degree_kernel(const int* __restrict__ src, const int* __restrict__ dst,
                              float* __restrict__ odeg, float* __restrict__ ideg, int E) {
  int e = blockIdx.x * blockDim.x + threadIdx.x;
  if (e < E) {
    atomicAdd(&odeg[src[e]], 1.0f);
    atomicAdd(&ideg[dst[e]], 1.0f);
  }
}

// in-place deg -> rsqrt(max(deg,1))
__global__ void invsqrt_kernel(float* __restrict__ a, float* __restrict__ b, int n) {
  int i = blockIdx.x * blockDim.x + threadIdx.x;
  if (i < n) {
    a[i] = rsqrtf(fmaxf(a[i], 1.0f));
    b[i] = rsqrtf(fmaxf(b[i], 1.0f));
  }
}

// W[k][n] (f32, K x Ncols row-major) -> Wt[n][k] (f16, transposed)
__global__ void wconv_kernel(const float* __restrict__ W, _Float16* __restrict__ Wt,
                             int K, int Ncols) {
  int i = blockIdx.x * blockDim.x + threadIdx.x;
  if (i < K * Ncols) {
    int k = i / Ncols, n = i % Ncols;
    Wt[(size_t)n * K + k] = (_Float16)W[i];
  }
}

// agg[dst[e]][:] += h[src[e]][:] * inv_out[src[e]]   (32 lanes/edge, float4/lane)
__global__ void scatter_kernel(const float* __restrict__ h, const int* __restrict__ src,
                               const int* __restrict__ dst, const float* __restrict__ inv_out,
                               float* __restrict__ agg, int E) {
  int t = blockIdx.x * blockDim.x + threadIdx.x;
  int e = t >> 5;
  if (e >= E) return;
  int lane = t & 31;
  int s = src[e], d = dst[e];
  float sc = inv_out[s];
  const float4 v = *(const float4*)(&h[(size_t)s * D + lane * 4]);
  float* base = &agg[(size_t)d * D + lane * 4];
  atomicAdd(base + 0, v.x * sc);
  atomicAdd(base + 1, v.y * sc);
  atomicAdd(base + 2, v.z * sc);
  atomicAdd(base + 3, v.w * sc);
}

// Out[M x ncols] = act( (A * rowscale) @ Wt^T + bias ), f16 WMMA, f32 accum.
// One 16x16 tile per wave; blockDim.x = (ncols/16)*32; grid.x = M/16.
__global__ void __launch_bounds__(256)
gemm_wmma(const float* __restrict__ A, const float* __restrict__ rowscale,
          const _Float16* __restrict__ Wt, const float* __restrict__ bias,
          float* __restrict__ Out, int K, int ncols, int relu) {
  const int lane = threadIdx.x & 31;
  const int wave = threadIdx.x >> 5;
  const int g    = lane >> 4;          // half-wave: selects K sub-run
  const int mn   = lane & 15;          // A row within tile / B column within tile
  const int row  = blockIdx.x * 16 + mn;
  const int col0 = wave * 16;

  const float s = rowscale ? rowscale[row] : 1.0f;
  v8f acc = {};

  const int nkt = K >> 5;              // K tiles of 32
  for (int kt = 0; kt < nkt; ++kt) {
    const int kb = kt << 5;
    // --- A fragment (16-bit A 16x32 layout): lane m+16g holds K = 8g..8g+7, 16+8g..16+8g+7
    float fa[16];
    const float* arow = A + (size_t)row * K + kb + 8 * g;
    *(float4*)&fa[0]  = *(const float4*)(arow);
    *(float4*)&fa[4]  = *(const float4*)(arow + 4);
    *(float4*)&fa[8]  = *(const float4*)(arow + 16);
    *(float4*)&fa[12] = *(const float4*)(arow + 20);
    v16h af;
#pragma unroll
    for (int i = 0; i < 16; ++i) af[i] = (_Float16)(fa[i] * s);
    // --- B fragment: lane n+16g holds W[K, col0+n] for same K runs; Wt is [n][k]
    const _Float16* wrow = Wt + (size_t)(col0 + mn) * K + kb + 8 * g;
    h8 blo = *(const h8*)(wrow);
    h8 bhi = *(const h8*)(wrow + 16);
    v16h bf = __builtin_shufflevector(blo, bhi, 0,1,2,3,4,5,6,7,8,9,10,11,12,13,14,15);
    acc = __builtin_amdgcn_wmma_f32_16x16x32_f16(false, af, false, bf,
                                                 (short)0, acc, false, false);
  }

  // C/D layout: VGPR r, lane n+16g -> element (M = r + 8g, N = n)
  const float bv = bias[col0 + mn];
#pragma unroll
  for (int r = 0; r < 8; ++r) {
    float v = acc[r] + bv;
    if (relu) v = fmaxf(v, 0.0f);
    Out[(size_t)(blockIdx.x * 16 + r + 8 * g) * ncols + col0 + mn] = v;
  }
}

// pooling
__global__ void pool_count_kernel(const int* __restrict__ gid, float* __restrict__ cnt, int n) {
  int i = blockIdx.x * blockDim.x + threadIdx.x;
  if (i < n) atomicAdd(&cnt[gid[i]], 1.0f);
}

__global__ void pool_sum_kernel(const float* __restrict__ x, const int* __restrict__ gid,
                                float* __restrict__ sums, int n) {
  int t = blockIdx.x * blockDim.x + threadIdx.x;
  int node = t >> 5;
  if (node >= n) return;
  int lane = t & 31;
  int gph = gid[node];
  const float4 v = *(const float4*)(&x[(size_t)node * D + lane * 4]);
  float* b = &sums[(size_t)gph * D + lane * 4];
  atomicAdd(b + 0, v.x);
  atomicAdd(b + 1, v.y);
  atomicAdd(b + 2, v.z);
  atomicAdd(b + 3, v.w);
}

__global__ void pool_div_kernel(const float* __restrict__ sums, const float* __restrict__ cnt,
                                float* __restrict__ hg) {
  int i = blockIdx.x * blockDim.x + threadIdx.x;
  if (i < NG * D) hg[i] = sums[i] / fmaxf(cnt[i / D], 1.0f);
}

// final 128 -> 10 (tiny, scalar)
__global__ void mlp_final_kernel(const float* __restrict__ t, const float* __restrict__ Wc2,
                                 const float* __restrict__ bc2, float* __restrict__ out) {
  int i = blockIdx.x * blockDim.x + threadIdx.x;
  if (i < NG * NC) {
    int gph = i / NC, c = i % NC;
    float acc = bc2[c];
#pragma unroll 4
    for (int k = 0; k < D; ++k) acc += t[(size_t)gph * D + k] * Wc2[(size_t)k * NC + c];
    out[i] = acc;
  }
}

extern "C" void kernel_launch(void* const* d_in, const int* in_sizes, int n_in,
                              void* d_out, int out_size, void* d_ws, size_t ws_size,
                              hipStream_t stream) {
  (void)in_sizes; (void)n_in; (void)out_size; (void)ws_size;

  const float* h_in = (const float*)d_in[0];
  const int*   src  = (const int*)d_in[1];
  const int*   dst  = (const int*)d_in[2];
  const int*   gid  = (const int*)d_in[3];
  const float* Wls[4] = {(const float*)d_in[4], (const float*)d_in[6],
                         (const float*)d_in[8], (const float*)d_in[10]};
  const float* bls[4] = {(const float*)d_in[5], (const float*)d_in[7],
                         (const float*)d_in[9], (const float*)d_in[11]};
  const float* Wc1 = (const float*)d_in[12];
  const float* bc1 = (const float*)d_in[13];
  const float* Wc2 = (const float*)d_in[14];
  const float* bc2 = (const float*)d_in[15];
  float* out = (float*)d_out;

  // ---- carve workspace ----
  char* ws = (char*)d_ws;
  size_t off = 0;
  auto carve = [&](size_t bytes) -> char* {
    char* p = ws + off;
    off = (off + bytes + 255) & ~(size_t)255;
    return p;
  };
  float*    inv_out = (float*)carve((size_t)NN * 4);
  float*    inv_in  = (float*)carve((size_t)NN * 4);
  float*    agg     = (float*)carve((size_t)NN * D * 4);
  float*    hbuf    = (float*)carve((size_t)NN * D * 4);
  _Float16* wt      = (_Float16*)carve((size_t)5 * D * D * 2); // 4 layers + Wc1
  float*    psum    = (float*)carve((size_t)NG * D * 4);
  float*    pcnt    = (float*)carve((size_t)NG * 4);
  float*    hg      = (float*)carve((size_t)NG * D * 4);
  float*    tbuf    = (float*)carve((size_t)NG * D * 4);

  const int TB = 256;

  // degrees -> inverse sqrt norms
  zero_f32<<<dim3((NN + TB - 1) / TB), TB, 0, stream>>>(inv_out, (size_t)NN);
  zero_f32<<<dim3((NN + TB - 1) / TB), TB, 0, stream>>>(inv_in, (size_t)NN);
  degree_kernel<<<dim3((NE + TB - 1) / TB), TB, 0, stream>>>(src, dst, inv_out, inv_in, NE);
  invsqrt_kernel<<<dim3((NN + TB - 1) / TB), TB, 0, stream>>>(inv_out, inv_in, NN);

  // convert all weights to transposed f16 once per call
  for (int l = 0; l < 4; ++l)
    wconv_kernel<<<dim3((D * D + TB - 1) / TB), TB, 0, stream>>>(Wls[l], wt + (size_t)l * D * D, D, D);
  wconv_kernel<<<dim3((D * D + TB - 1) / TB), TB, 0, stream>>>(Wc1, wt + (size_t)4 * D * D, D, D);

  // ---- 4 graph-conv layers ----
  const dim3 gemm_grid(NN / 16);   // 6250, N divisible by 16
  for (int l = 0; l < 4; ++l) {
    const float* hcur = (l == 0) ? h_in : hbuf;
    zero_f32<<<dim3(4096), TB, 0, stream>>>(agg, (size_t)NN * D);
    scatter_kernel<<<dim3((size_t)NE * 32 / TB), TB, 0, stream>>>(hcur, src, dst, inv_out, agg, NE);
    gemm_wmma<<<gemm_grid, 256, 0, stream>>>(agg, inv_in, wt + (size_t)l * D * D,
                                             bls[l], hbuf, D, D, /*relu=*/1);
  }

  // ---- mean pooling over graphs ----
  zero_f32<<<dim3((NG * D + TB - 1) / TB), TB, 0, stream>>>(psum, (size_t)NG * D);
  zero_f32<<<dim3(1), TB, 0, stream>>>(pcnt, (size_t)NG);
  pool_count_kernel<<<dim3((NN + TB - 1) / TB), TB, 0, stream>>>(gid, pcnt, NN);
  pool_sum_kernel<<<dim3((size_t)NN * 32 / TB + 1), TB, 0, stream>>>(hbuf, gid, psum, NN);
  pool_div_kernel<<<dim3((NG * D + TB - 1) / TB), TB, 0, stream>>>(psum, pcnt, hg);

  // ---- classifier head ----
  gemm_wmma<<<dim3(NG / 16), 256, 0, stream>>>(hg, nullptr, wt + (size_t)4 * D * D,
                                               bc1, tbuf, D, D, /*relu=*/1);
  mlp_final_kernel<<<dim3((NG * NC + TB - 1) / TB), TB, 0, stream>>>(tbuf, Wc2, bc2, out);
}
